// Attention_601295422127
// MI455X (gfx1250) — matrix operand order repaired and checked by reading, not compiled, not run on hardware
//
#include <hip/hip_runtime.h>
#include <hip/hip_bf16.h>

typedef __attribute__((ext_vector_type(16))) __bf16 v16bf;
typedef __attribute__((ext_vector_type(8)))  float  v8f;

#define WMMA_BF16(a, b, c) \
  __builtin_amdgcn_wmma_f32_16x16x32_bf16(false, (a), false, (b), (short)0, (c), false, false)

// ---------------------------------------------------------------------------
// Kernel 0: convert weights to bf16 in WMMA-friendly layouts.
//   Wq/Wk/Wv [H][512][64] f32  ->  W*T [H][64][512] bf16  (B-frag contiguous)
//   Wo [512][512] f32 -> bf16 as-is (reduction over 'in' is contiguous)
// All weights stored UNSCALED; the 1/sqrt(E) softmax scale is applied as
// outscale=0.125 in the q projection epilogue: (q@Wq + bq) * 0.125.
// ---------------------------------------------------------------------------
__global__ __launch_bounds__(256) void prep_weights(
    const float* __restrict__ Wq, const float* __restrict__ Wk,
    const float* __restrict__ Wv, const float* __restrict__ Wo,
    __bf16* __restrict__ WqT, __bf16* __restrict__ WkT,
    __bf16* __restrict__ WvT, __bf16* __restrict__ WoB) {
  int idx = blockIdx.x * blockDim.x + threadIdx.x;   // 0 .. 262143
  int e = idx & 63;
  int d = (idx >> 6) & 511;
  int h = idx >> 15;
  int src = (h * 512 + d) * 64 + e;                  // [h][d][e]
  int dst = (h * 64 + e) * 512 + d;                  // [h][e][d]
  WqT[dst] = (__bf16)Wq[src];
  WkT[dst] = (__bf16)Wk[src];
  WvT[dst] = (__bf16)Wv[src];
  WoB[idx] = (__bf16)Wo[idx];                        // [o][d] row-major
}

// ---------------------------------------------------------------------------
// Kernel 1: per-head projection  Y[b,h,s,:] = (X[b,s,:] @ W[h] + bias[h])*sc
// X: [8192,512] f32, WT: [H][64][512] bf16, Y bf16 (optionally transposed
// [B,H,E,S] for the V projection). grid = (64, H), block = 256 (8 waves),
// wave computes a 16x64 output tile via 4 f32 accumulators, K-loop step 32.
// ---------------------------------------------------------------------------
__global__ __launch_bounds__(256) void proj_kernel(
    const float* __restrict__ X, const __bf16* __restrict__ WT,
    const float* __restrict__ bias, __bf16* __restrict__ Y,
    float outscale, int transposed) {
  const int h = blockIdx.y;
  const int lane = threadIdx.x & 31;
  const int wave = threadIdx.x >> 5;
  const int l15 = lane & 15;
  const int lhi = lane >> 4;
  const int rowBase = blockIdx.x * 128 + wave * 16;

  const float* xrow = X + (long)(rowBase + l15) * 512;
  v8f acc[4];
#pragma unroll
  for (int eg = 0; eg < 4; ++eg) acc[eg] = (v8f){};

  for (int k = 0; k < 512; k += 32) {
    // A fragment: 16x32 bf16 from fp32 input
    v16bf a;
    const int klo = k + (lhi ? 16 : 0);
#pragma unroll
    for (int i = 0; i < 8; ++i) a[i] = (__bf16)xrow[klo + i];
#pragma unroll
    for (int i = 0; i < 8; ++i) a[8 + i] = (__bf16)xrow[klo + 8 + i];

    const int dbase = k + (lhi ? 16 : 0);
#pragma unroll
    for (int eg = 0; eg < 4; ++eg) {
      const int e = eg * 16 + l15;
      v16bf b = *(const v16bf*)(WT + ((long)(h * 64 + e) * 512 + dbase));
      acc[eg] = WMMA_BF16(a, b, acc[eg]);
    }
  }

  // epilogue: (+ bias) * outscale, scatter per C-layout
#pragma unroll
  for (int eg = 0; eg < 4; ++eg) {
    const int e = eg * 16 + l15;
    const float bv = bias[h * 64 + e];
#pragma unroll
    for (int r = 0; r < 8; ++r) {
      const int M = r + (lhi ? 8 : 0);
      const int row = rowBase + M;                    // global row in [0,8192)
      const int b2 = row >> 11;                       // batch
      const int s  = row & 2047;                      // seq pos
      const float val = (acc[eg][r] + bv) * outscale;
      long idx;
      if (transposed)  idx = ((long)(b2 * 8 + h) * 64 + e) * 2048 + s;   // [B,H,E,S]
      else             idx = ((long)(b2 * 8 + h) * 2048 + s) * 64 + e;   // [B,H,S,E]
      Y[idx] = (__bf16)val;
    }
  }
}

// ---------------------------------------------------------------------------
// Kernel 2: flash attention per (b,h). grid = (Sq/128, B*H), block 256.
// Each wave: 16 query rows, streams 32-key tiles. Scores via 4 WMMAs,
// online softmax (shfl_xor row reductions), P through LDS (C->A transpose),
// P@V via 4 WMMAs into 16x64 f32 accumulator. Output -> concat [B,S,H*E] bf16.
// ---------------------------------------------------------------------------
__global__ __launch_bounds__(256) void attn_kernel(
    const __bf16* __restrict__ qh, const __bf16* __restrict__ kh,
    const __bf16* __restrict__ vhT, __bf16* __restrict__ concat) {
  __shared__ __bf16 ptile[8][16][34];   // per-wave 16x32 P tile (+pad)

  const int bh = blockIdx.y;            // 0..31 = b*8 + h
  const int lane = threadIdx.x & 31;
  const int wave = threadIdx.x >> 5;
  const int l15 = lane & 15;
  const int lhi = lane >> 4;
  const int qBase = blockIdx.x * 128 + wave * 16;

  const __bf16* qb = qh  + (long)bh * 2048 * 64;
  const __bf16* kb = kh  + (long)bh * 2048 * 64;
  const __bf16* vb = vhT + (long)bh * 64 * 2048;

  // Q A-fragments for K-steps 0 and 32 (held in registers all loop long)
  v16bf aq0, aq1;
  {
    const __bf16* qrow = qb + (long)(qBase + l15) * 64;
    const int klo = lhi ? 16 : 0;
#pragma unroll
    for (int i = 0; i < 8; ++i) {
      aq0[i]     = qrow[klo + i];
      aq0[8 + i] = qrow[klo + 8 + i];
      aq1[i]     = qrow[32 + klo + i];
      aq1[8 + i] = qrow[32 + klo + 8 + i];
    }
  }

  float m[8], lsum[8];
  v8f accO[4];
#pragma unroll
  for (int r = 0; r < 8; ++r) { m[r] = -3.0e38f; lsum[r] = 0.0f; }
#pragma unroll
  for (int eg = 0; eg < 4; ++eg) accO[eg] = (v8f){};

  for (int t = 0; t < 2048; t += 32) {
    // ---- scores S = Qs @ K^T for 32 keys (two 16-col fragments) ----
    v8f sfrag[2];
#pragma unroll
    for (int ng = 0; ng < 2; ++ng) {
      const int key = t + ng * 16 + l15;
      const __bf16* krow = kb + (long)key * 64 + (lhi ? 16 : 0);
      v16bf b0 = *(const v16bf*)(krow);
      v16bf b1 = *(const v16bf*)(krow + 32);
      v8f z = (v8f){};
      z = WMMA_BF16(aq0, b0, z);
      z = WMMA_BF16(aq1, b1, z);
      sfrag[ng] = z;
    }

    // ---- online softmax: row max across 32 keys (16-lane groups) ----
    float alpha[8];
#pragma unroll
    for (int r = 0; r < 8; ++r) {
      float mx = fmaxf(sfrag[0][r], sfrag[1][r]);
      mx = fmaxf(mx, __shfl_xor(mx, 1, 32));
      mx = fmaxf(mx, __shfl_xor(mx, 2, 32));
      mx = fmaxf(mx, __shfl_xor(mx, 4, 32));
      mx = fmaxf(mx, __shfl_xor(mx, 8, 32));
      const float mn = fmaxf(m[r], mx);
      alpha[r] = __expf(m[r] - mn);
      m[r] = mn;
    }

    // ---- P = exp(S - m), partial row sums, stage P into LDS ----
#pragma unroll
    for (int r = 0; r < 8; ++r) {
      const int Mr = r + (lhi ? 8 : 0);
      const float p0 = __expf(sfrag[0][r] - m[r]);
      const float p1 = __expf(sfrag[1][r] - m[r]);
      lsum[r] = alpha[r] * lsum[r] + p0 + p1;
      ptile[wave][Mr][l15]      = (__bf16)p0;
      ptile[wave][Mr][16 + l15] = (__bf16)p1;
#pragma unroll
      for (int eg = 0; eg < 4; ++eg) accO[eg][r] *= alpha[r];
    }
    __syncthreads();   // order LDS stores before A-layout reads

    // ---- reload P in A-fragment layout ----
    v16bf pa;
    {
      const int klo = lhi ? 16 : 0;
#pragma unroll
      for (int i = 0; i < 8; ++i) {
        pa[i]     = ptile[wave][l15][klo + i];
        pa[8 + i] = ptile[wave][l15][klo + 8 + i];
      }
    }
    __syncthreads();   // protect tile from next-iteration overwrite

    // ---- O += P @ V  (V transposed: contiguous B-fragment loads) ----
#pragma unroll
    for (int eg = 0; eg < 4; ++eg) {
      const int e = eg * 16 + l15;
      v16bf bv = *(const v16bf*)(vb + (long)e * 2048 + t + (lhi ? 16 : 0));
      accO[eg] = WMMA_BF16(pa, bv, accO[eg]);
    }
  }

  // ---- epilogue: full row sums, normalize, write concat [B,S,H*E] ----
  const int b2 = bh >> 3;
  const int h  = bh & 7;
#pragma unroll
  for (int r = 0; r < 8; ++r) {
    float ls = lsum[r];
    ls += __shfl_xor(ls, 1, 32);
    ls += __shfl_xor(ls, 2, 32);
    ls += __shfl_xor(ls, 4, 32);
    ls += __shfl_xor(ls, 8, 32);
    const float inv = 1.0f / ls;
    const int M = r + (lhi ? 8 : 0);
    const long sq = qBase + M;
#pragma unroll
    for (int eg = 0; eg < 4; ++eg) {
      const long idx = ((long)b2 * 2048 + sq) * 512 + h * 64 + eg * 16 + l15;
      concat[idx] = (__bf16)(accO[eg][r] * inv);
    }
  }
}

// ---------------------------------------------------------------------------
// Kernel 3: output projection  y = concat @ Wo^T   (f32 output)
// grid = (8192/128, 512/64), block 256; wave computes 16x64 tile.
// ---------------------------------------------------------------------------
__global__ __launch_bounds__(256) void out_gemm(
    const __bf16* __restrict__ concat, const __bf16* __restrict__ WoB,
    float* __restrict__ y) {
  const int lane = threadIdx.x & 31;
  const int wave = threadIdx.x >> 5;
  const int l15 = lane & 15;
  const int lhi = lane >> 4;
  const int rowBase = blockIdx.x * 128 + wave * 16;
  const int oBase = blockIdx.y * 64;

  const __bf16* xrow = concat + (long)(rowBase + l15) * 512;
  v8f acc[4];
#pragma unroll
  for (int og = 0; og < 4; ++og) acc[og] = (v8f){};

  for (int k = 0; k < 512; k += 32) {
    v16bf a;
    const int klo = k + (lhi ? 16 : 0);
#pragma unroll
    for (int i = 0; i < 8; ++i) {
      a[i]     = xrow[klo + i];
      a[8 + i] = xrow[klo + 8 + i];
    }
    const int dbase = k + (lhi ? 16 : 0);
#pragma unroll
    for (int og = 0; og < 4; ++og) {
      const int o = oBase + og * 16 + l15;
      v16bf b = *(const v16bf*)(WoB + (long)o * 512 + dbase);
      acc[og] = WMMA_BF16(a, b, acc[og]);
    }
  }
#pragma unroll
  for (int og = 0; og < 4; ++og) {
    const int o = oBase + og * 16 + l15;
#pragma unroll
    for (int r = 0; r < 8; ++r) {
      const int row = rowBase + r + (lhi ? 8 : 0);
      y[(long)row * 512 + o] = acc[og][r];
    }
  }
}

// ---------------------------------------------------------------------------
extern "C" void kernel_launch(void* const* d_in, const int* in_sizes, int n_in,
                              void* d_out, int out_size, void* d_ws, size_t ws_size,
                              hipStream_t stream) {
  const float* q  = (const float*)d_in[0];
  const float* k  = (const float*)d_in[1];
  const float* v  = (const float*)d_in[2];
  const float* Wq = (const float*)d_in[3];
  const float* bq = (const float*)d_in[4];
  const float* Wk = (const float*)d_in[5];
  const float* bk = (const float*)d_in[6];
  const float* Wv = (const float*)d_in[7];
  const float* bv = (const float*)d_in[8];
  const float* Wo = (const float*)d_in[9];
  float* out = (float*)d_out;

  // workspace layout (bf16 tensors), ~35.6 MB total
  char* ws = (char*)d_ws;
  const size_t BHSE = (size_t)4 * 8 * 2048 * 64 * sizeof(__bf16);   // 8 MiB
  __bf16* qh     = (__bf16*)(ws);              // [B,H,S,E]   (pre-scaled 1/8)
  __bf16* kh     = (__bf16*)(ws + BHSE);       // [B,H,S,E]
  __bf16* vhT    = (__bf16*)(ws + 2 * BHSE);   // [B,H,E,S]
  __bf16* concat = (__bf16*)(ws + 3 * BHSE);   // [B,S,H*E]
  char* wsw = ws + 4 * BHSE;
  const size_t WSZ = (size_t)8 * 512 * 64 * sizeof(__bf16);         // 512 KiB
  __bf16* WqT = (__bf16*)(wsw);
  __bf16* WkT = (__bf16*)(wsw + WSZ);
  __bf16* WvT = (__bf16*)(wsw + 2 * WSZ);
  __bf16* WoB = (__bf16*)(wsw + 3 * WSZ);      // [512][512]

  prep_weights<<<1024, 256, 0, stream>>>(Wq, Wk, Wv, Wo, WqT, WkT, WvT, WoB);

  // projections: 8192 rows / 128 per block, one head per blockIdx.y.
  // q path applies outscale = 1/sqrt(E) = 0.125 to (q@Wq + bq), which is
  // mathematically identical to dividing the attention scores by sqrt(E).
  proj_kernel<<<dim3(64, 8), 256, 0, stream>>>(q, WqT, bq, qh,  0.125f, 0);
  proj_kernel<<<dim3(64, 8), 256, 0, stream>>>(k, WkT, bk, kh,  1.0f,   0);
  proj_kernel<<<dim3(64, 8), 256, 0, stream>>>(v, WvT, bv, vhT, 1.0f,   1);

  // flash attention: 16 query tiles x 32 (b,h) pairs
  attn_kernel<<<dim3(16, 32), 256, 0, stream>>>(qh, kh, vhT, concat);

  // output projection
  out_gemm<<<dim3(64, 8), 256, 0, stream>>>(concat, WoB, out);
}
